// MANN_87308095193290
// MI455X (gfx1250) — matrix operand order repaired and checked by reading, not compiled, not run on hardware
//
#include <hip/hip_runtime.h>
#include <hip/hip_bf16.h>
#include <math.h>

// ---------------------------------------------------------------------------
// MoE "MANN" forward, fused, WMMA f16 path for gfx1250 (MI455X).
// Blocking: 64 samples/block, 16 waves = 8 experts x 2 N-halves.
// Each wave: 4 M-subtiles x 4 N-tiles -> B-fragment reuse x4 (cuts L2 weight
// traffic ~4x vs one-subtile-per-wave), f32 accum, gate-mix via ds_add_f32
// into 2 LDS buffers (e&1) to halve atomic contention.
// ---------------------------------------------------------------------------

#define NCHAR 156
#define NGATE 12
#define DIN   168      // NCHAR + NGATE
#define NEXP  8
#define HID   128
#define OUTD  96
#define KP0   160      // K of layer0 padded to multiple of 32
#define ASTR  168      // LDS A-tile row stride in f16 elems (16B aligned, conflict-friendly)
#define MROWS 64       // samples per block
#define MSUB  4        // 16-row subtiles per wave

typedef _Float16 v16h __attribute__((ext_vector_type(16)));
typedef _Float16 v8h  __attribute__((ext_vector_type(8)));
typedef float    v8f  __attribute__((ext_vector_type(8)));

__device__ __forceinline__ float eluf(float x) {
    return x > 0.0f ? x : expm1f(x);
}

// ---------------------------------------------------------------------------
// Weight repack: f32 (E,K,N) -> f16 WMMA B-fragments.
// Fragment (e,kc,nt) = 32 lanes x 16 halfs, lane l covers column nt*16+(l&15),
// K values kc*32 + (l>=16 ? 16 : 0) + j  (j = 0..15).  K zero-padded.
// ---------------------------------------------------------------------------
__global__ void pack_kernel(const float* __restrict__ W, _Float16* __restrict__ dst,
                            int Kact, int N, int KC, int NT, int total)
{
    int t = blockIdx.x * 256 + threadIdx.x;
    if (t >= total) return;
    int j    = t & 15;
    int lane = (t >> 4) & 31;
    int frag = t >> 9;
    int nt   = frag % NT;
    int rest = frag / NT;
    int kc   = rest % KC;
    int e    = rest / KC;
    int n = nt * 16 + (lane & 15);
    int k = kc * 32 + ((lane >> 4) << 4) + j;
    float v = (k < Kact) ? W[((size_t)e * Kact + k) * N + n] : 0.0f;
    dst[t] = (_Float16)v;
}

// ---------------------------------------------------------------------------
// Gating MLP: phase(12) -> 32 -> 32 -> 8, ELU, softmax.  One thread / sample.
// ---------------------------------------------------------------------------
__global__ __launch_bounds__(256) void gate_kernel(
    const float* __restrict__ X, const float* __restrict__ xnorm,
    const float* __restrict__ g0w, const float* __restrict__ g0b,
    const float* __restrict__ g1w, const float* __restrict__ g1b,
    const float* __restrict__ g2w, const float* __restrict__ g2b,
    float* __restrict__ gate, int B)
{
    __shared__ float s0[NGATE * 32];
    __shared__ float s1[32 * 32];
    __shared__ float s2[32 * NEXP];
    __shared__ float sb0[32], sb1[32], sb2[NEXP];

    int tid = threadIdx.x;
    for (int i = tid; i < NGATE * 32; i += 256) s0[i] = g0w[i];
    for (int i = tid; i < 32 * 32;   i += 256) s1[i] = g1w[i];
    for (int i = tid; i < 32 * NEXP; i += 256) s2[i] = g2w[i];
    if (tid < 32) { sb0[tid] = g0b[tid]; sb1[tid] = g1b[tid]; }
    if (tid < NEXP) sb2[tid] = g2b[tid];
    __syncthreads();

    int b = blockIdx.x * 256 + tid;
    if (b >= B) return;

    float ph[NGATE];
    #pragma unroll
    for (int j = 0; j < NGATE; ++j) {
        int c = NCHAR + j;
        ph[j] = (X[(size_t)b * DIN + c] - xnorm[c]) / xnorm[DIN + c];
    }
    float h1[32];
    #pragma unroll
    for (int o = 0; o < 32; ++o) {
        float s = sb0[o];
        #pragma unroll
        for (int j = 0; j < NGATE; ++j) s = fmaf(ph[j], s0[j * 32 + o], s);
        h1[o] = eluf(s);
    }
    float h2[32];
    #pragma unroll
    for (int o = 0; o < 32; ++o) {
        float s = sb1[o];
        #pragma unroll
        for (int j = 0; j < 32; ++j) s = fmaf(h1[j], s1[j * 32 + o], s);
        h2[o] = eluf(s);
    }
    float lg[NEXP];
    #pragma unroll
    for (int o = 0; o < NEXP; ++o) {
        float s = sb2[o];
        #pragma unroll
        for (int j = 0; j < 32; ++j) s = fmaf(h2[j], s2[j * NEXP + o], s);
        lg[o] = s;
    }
    float m = lg[0];
    #pragma unroll
    for (int o = 1; o < NEXP; ++o) m = fmaxf(m, lg[o]);
    float den = 0.0f;
    #pragma unroll
    for (int o = 0; o < NEXP; ++o) { lg[o] = expf(lg[o] - m); den += lg[o]; }
    float inv = 1.0f / den;
    #pragma unroll
    for (int o = 0; o < NEXP; ++o) gate[(size_t)b * NEXP + o] = lg[o] * inv;
}

// ---------------------------------------------------------------------------
// One expert layer, one wave: 4 subtiles (64 rows) x NTL n-tiles of one expert.
// acc[m][nt] = A_m(16xK) @ Wp[e](Kx16) via WMMA; then gate-weighted mix
// (+bias) into H0/H1 (selected by e&1) with ds_add_f32.
// NTOT = total n-tiles of the layer; this wave owns ntiles nh*NTL..nh*NTL+NTL-1.
// ---------------------------------------------------------------------------
template <int KC, int NTL, int NTOT>
__device__ __forceinline__ void expert_layer(
    const _Float16* __restrict__ Wpack, const float* __restrict__ bias,
    int e, int nh, int lane,
    const _Float16* A, float* H0, float* H1, const float* gsh)
{
    const int hi = lane >> 4;
    const int lc = lane & 15;

    v8f acc[MSUB][NTL] = {};
    const v16h* Wp = (const v16h*)Wpack;

    #pragma unroll
    for (int kc = 0; kc < KC; ++kc) {
        // A fragments (16x32 f16): lane holds row lc, K halves
        // {0..15} (lanes 0-15) / {16..31} (lanes 16-31)
        const int kb = kc * 32 + hi * 16;
        v16h af[MSUB];
        #pragma unroll
        for (int m = 0; m < MSUB; ++m) {
            const int r = m * 16 + lc;
            v8h a0 = *(const v8h*)(A + r * ASTR + kb);
            v8h a1 = *(const v8h*)(A + r * ASTR + kb + 8);
            af[m] = __builtin_shufflevector(a0, a1, 0, 1, 2, 3, 4, 5, 6, 7,
                                                    8, 9, 10, 11, 12, 13, 14, 15);
        }
        if (kc + 1 < KC) {
            __builtin_prefetch(
                (const void*)&Wp[(size_t)((e * KC + kc + 1) * NTOT + nh * NTL) * 32 + lane],
                0, 3);
        }
        #pragma unroll
        for (int nt = 0; nt < NTL; ++nt) {
            v16h bf = Wp[(size_t)((e * KC + kc) * NTOT + nh * NTL + nt) * 32 + lane];
            #pragma unroll
            for (int m = 0; m < MSUB; ++m) {
                acc[m][nt] = __builtin_amdgcn_wmma_f32_16x16x32_f16(
                    false, af[m], false, bf, (short)0, acc[m][nt], false, false);
            }
        }
    }

    float bv[NTL];
    #pragma unroll
    for (int nt = 0; nt < NTL; ++nt) bv[nt] = bias[(nh * NTL + nt) * 16 + lc];

    float* Hsel = (e & 1) ? H1 : H0;
    #pragma unroll
    for (int m = 0; m < MSUB; ++m) {
        #pragma unroll
        for (int i = 0; i < 8; ++i) {
            const int row = m * 16 + i + hi * 8;
            const float g = gsh[row * NEXP + e];
            #pragma unroll
            for (int nt = 0; nt < NTL; ++nt) {
                const int col = (nh * NTL + nt) * 16 + lc;
                atomicAdd(&Hsel[row * HID + col], (acc[m][nt][i] + bv[nt]) * g);
            }
        }
    }
}

// ---------------------------------------------------------------------------
// Fused 3-layer MoE kernel: 64 samples/block, 16 waves = (expert, N-half)
// ---------------------------------------------------------------------------
__global__ __launch_bounds__(512) void moe_kernel(
    const float* __restrict__ X, const float* __restrict__ xnorm,
    const float* __restrict__ ynorm, const float* __restrict__ gate,
    const _Float16* __restrict__ W0p, const _Float16* __restrict__ W1p,
    const _Float16* __restrict__ W2p,
    const float* __restrict__ b0, const float* __restrict__ b1,
    const float* __restrict__ b2,
    float* __restrict__ out, int B)
{
    __shared__ _Float16 Atile[MROWS * ASTR];   // 21 KB activations (f16)
    __shared__ float    H0[MROWS * HID];       // 32 KB mixed f32 (experts e&1==0)
    __shared__ float    H1[MROWS * HID];       // 32 KB mixed f32 (experts e&1==1)
    __shared__ float    gsh[MROWS * NEXP];     // 2 KB gate weights

    const int tid  = threadIdx.x;
    const int lane = tid & 31;
    const int wv   = tid >> 5;
    const int e    = wv & 7;
    const int nh   = wv >> 3;                  // N-half: 0 or 1
    const int row0 = blockIdx.x * MROWS;

    // init: zero H0/H1, stage gate tile, stage normalized char input as f16
    for (int i = tid; i < MROWS * HID; i += 512) { H0[i] = 0.0f; H1[i] = 0.0f; }
    for (int i = tid; i < MROWS * NEXP; i += 512) {
        int r = i >> 3, g = i & 7;
        int b = min(row0 + r, B - 1);
        gsh[i] = gate[(size_t)b * NEXP + g];
    }
    for (int i = tid; i < MROWS * KP0; i += 512) {
        int r = i / KP0, c = i % KP0;
        int b = min(row0 + r, B - 1);
        float v = 0.0f;
        if (c < NCHAR) v = (X[(size_t)b * DIN + c] - xnorm[c]) / xnorm[DIN + c];
        Atile[r * ASTR + c] = (_Float16)v;
    }
    __syncthreads();

    // Layer 0: 156(pad 160) -> 128
    expert_layer<KP0 / 32, 4, HID / 16>(W0p, b0 + e * HID, e, nh, lane, Atile, H0, H1, gsh);
    __syncthreads();
    for (int i = tid; i < MROWS * HID; i += 512) {
        float v = eluf(H0[i] + H1[i]);
        Atile[(i >> 7) * ASTR + (i & 127)] = (_Float16)v;
        H0[i] = 0.0f; H1[i] = 0.0f;
    }
    __syncthreads();

    // Layer 1: 128 -> 128
    expert_layer<HID / 32, 4, HID / 16>(W1p, b1 + e * HID, e, nh, lane, Atile, H0, H1, gsh);
    __syncthreads();
    for (int i = tid; i < MROWS * HID; i += 512) {
        float v = eluf(H0[i] + H1[i]);
        Atile[(i >> 7) * ASTR + (i & 127)] = (_Float16)v;
        H0[i] = 0.0f; H1[i] = 0.0f;
    }
    __syncthreads();

    // Layer 2: 128 -> 96 (no activation)
    expert_layer<HID / 32, 3, OUTD / 16>(W2p, b2 + e * OUTD, e, nh, lane, Atile, H0, H1, gsh);
    __syncthreads();

    // de-normalize + store
    for (int i = tid; i < MROWS * OUTD; i += 512) {
        int r = i / OUTD, c = i % OUTD;
        int b = row0 + r;
        if (b < B)
            out[(size_t)b * OUTD + c] = (H0[r * HID + c] + H1[r * HID + c]) * ynorm[OUTD + c] + ynorm[c];
    }
}

// ---------------------------------------------------------------------------
extern "C" void kernel_launch(void* const* d_in, const int* in_sizes, int n_in,
                              void* d_out, int out_size, void* d_ws, size_t ws_size,
                              hipStream_t stream)
{
    (void)n_in; (void)out_size; (void)ws_size;

    const float* X     = (const float*)d_in[0];
    const float* xnorm = (const float*)d_in[1];
    const float* ynorm = (const float*)d_in[2];
    const float* g0w   = (const float*)d_in[3];
    const float* g0b   = (const float*)d_in[4];
    const float* g1w   = (const float*)d_in[5];
    const float* g1b   = (const float*)d_in[6];
    const float* g2w   = (const float*)d_in[7];
    const float* g2b   = (const float*)d_in[8];
    const float* W0    = (const float*)d_in[9];
    const float* b0    = (const float*)d_in[10];
    const float* W1    = (const float*)d_in[11];
    const float* b1    = (const float*)d_in[12];
    const float* W2    = (const float*)d_in[13];
    const float* b2    = (const float*)d_in[14];
    float* out = (float*)d_out;

    const int B = in_sizes[0] / DIN;

    // workspace carve-up
    float* gate_ws = (float*)d_ws;                                 // B*8 f32
    char*  wsbase  = (char*)d_ws + (size_t)B * NEXP * sizeof(float);
    _Float16* W0p = (_Float16*)wsbase;                             // 8*5*8*512 halfs
    _Float16* W1p = W0p + (size_t)NEXP * 5 * 8 * 512;              // 8*4*8*512
    _Float16* W2p = W1p + (size_t)NEXP * 4 * 8 * 512;              // 8*4*6*512

    // 1) repack weights to f16 WMMA B-fragments (tiny; weights then live in L2)
    {
        int t0 = NEXP * 5 * 8 * 512;
        pack_kernel<<<(t0 + 255) / 256, 256, 0, stream>>>(W0, W0p, NCHAR, HID, 5, 8, t0);
        int t1 = NEXP * 4 * 8 * 512;
        pack_kernel<<<(t1 + 255) / 256, 256, 0, stream>>>(W1, W1p, HID, HID, 4, 8, t1);
        int t2 = NEXP * 4 * 6 * 512;
        pack_kernel<<<(t2 + 255) / 256, 256, 0, stream>>>(W2, W2p, HID, OUTD, 4, 6, t2);
    }

    // 2) gating MLP -> softmax weights
    gate_kernel<<<(B + 255) / 256, 256, 0, stream>>>(
        X, xnorm, g0w, g0b, g1w, g1b, g2w, g2b, gate_ws, B);

    // 3) fused 3-layer mixture-of-experts (WMMA)
    moe_kernel<<<(B + MROWS - 1) / MROWS, 512, 0, stream>>>(
        X, xnorm, ynorm, gate_ws, W0p, W1p, W2p, b0, b1, b2, out, B);
}
